// RelationMemory_16192026706627
// MI455X (gfx1250) — matrix operand order repaired and checked by reading, not compiled, not run on hardware
//
#include <hip/hip_runtime.h>

typedef __attribute__((ext_vector_type(16))) _Float16 v16h;
typedef __attribute__((ext_vector_type(8)))  _Float16 v8h;
typedef __attribute__((ext_vector_type(8)))  float    v8f;
typedef __attribute__((ext_vector_type(4)))  float    v4f;

static constexpr int BSZ   = 128;
static constexpr int DIM   = 128;
static constexpr int DSIN  = 1024;
static constexpr int NNEG  = 16;
static constexpr int NPAIR = BSZ * BSZ;          // 16384
static constexpr int NMEM  = 100000;
static constexpr int WST   = 136;                // padded LDS stride (halves): 4-bank shift/row
static constexpr int TILE_H = 16 * WST;          // one 16x128 activation tile (halves)
static constexpr int WMAT_H = 128 * WST;         // one padded 128x128 weight matrix (halves)
static constexpr float INV_T = 1.0f / 0.07f;

#define ASYNC_W 1   // stage weights into LDS with global_load_async_to_lds_b128 (ASYNCcnt)

// ---------------- WMMA fragment helpers (gfx1250 16x16x32 f16) ----------------
// A (16x32 f16): lanes 0-15 row m=lane, K[0..7]+K[16..23]; lanes 16-31 row m=lane-16, K[8..15]+K[24..31]
__device__ __forceinline__ v16h load_a_frag(const _Float16* base, int ld, int kk) {
  int lane = threadIdx.x & 31;
  int m    = lane & 15;
  int hi   = lane >> 4;
  const _Float16* p = base + m * ld + kk * 32 + hi * 8;
  v8h a0 = *(const v8h*)(p);
  v8h a1 = *(const v8h*)(p + 16);
  v16h a;
#pragma unroll
  for (int q = 0; q < 8; ++q) { a[q] = a0[q]; a[8 + q] = a1[q]; }
  return a;
}

// B (32x16 f16), B(k,n) = W[n*ld + k] (W row-major [out,in]):
// lanes 0-15: col n=lane, K[0..7]+K[16..23]; lanes 16-31: col n=lane-16, K[8..15]+K[24..31]
__device__ __forceinline__ v16h load_b_frag(const _Float16* w, int ld, int tn, int kk) {
  int lane = threadIdx.x & 31;
  int n = tn * 16 + (lane & 15);
  int k = kk * 32 + (lane >> 4) * 8;
  const _Float16* p = w + n * ld + k;
  v8h b0 = *(const v8h*)(p);
  v8h b1 = *(const v8h*)(p + 16);
  v16h b;
#pragma unroll
  for (int q = 0; q < 8; ++q) { b[q] = b0[q]; b[8 + q] = b1[q]; }
  return b;
}

// [16 x 32*KSTEPS] (A) x [32*KSTEPS x 128] (B = W^T) -> acc[8] (8 n-tiles of 16x16 f32)
template <int KSTEPS>
__device__ __forceinline__ void gemm16(const _Float16* A, int lda,
                                       const _Float16* W, int ldb, v8f acc[8]) {
#pragma unroll
  for (int kk = 0; kk < KSTEPS; ++kk) {
    v16h a = load_a_frag(A, lda, kk);
#pragma unroll
    for (int tn = 0; tn < 8; ++tn) {
      v16h b = load_b_frag(W, ldb, tn, kk);
      acc[tn] = __builtin_amdgcn_wmma_f32_16x16x32_f16(
          false, a, false, b, (short)0, acc[tn], false, false);
    }
  }
}

// ---------------- async weight staging: global f16 [128x128] -> padded LDS ----------------
__device__ __forceinline__ void lds_copy_weight(_Float16* dst, const _Float16* __restrict__ src) {
#if ASYNC_W
  // 128 rows x 256B; 16B chunks, DMA'd LDS-side with no VGPR round trip (ASYNCcnt).
  unsigned int base = (unsigned int)(unsigned long long)dst;  // low 32 bits of generic = LDS offset
  for (int c = threadIdx.x; c < 128 * 16; c += 256) {
    int row = c >> 4, seg = c & 15;
    unsigned int loff = base + row * (WST * 2) + seg * 16;
    const char* gp = (const char*)src + row * 256 + seg * 16;
    asm volatile("global_load_async_to_lds_b128 %0, %1, off"
                 :: "v"(loff), "v"(gp) : "memory");
  }
#else
  for (int c = threadIdx.x; c < 128 * 128; c += 256) {
    int r = c >> 7, col = c & 127;
    dst[r * WST + col] = src[c];
  }
#endif
}

__device__ __forceinline__ void weights_ready() {
#if ASYNC_W
  asm volatile("s_wait_asynccnt 0x0" ::: "memory");
#endif
  __syncthreads();
}

// ---------------- elementwise f32 -> f16 convert ----------------
__global__ void k_cvt(const float* __restrict__ src, _Float16* __restrict__ dst, int n) {
  int i = blockIdx.x * blockDim.x + threadIdx.x;
  if (i < n) dst[i] = (_Float16)src[i];
}

// ---------------- se/te embed: [128x1024] x [1024x128] ----------------
__global__ void k_embed(const _Float16* __restrict__ s16, const _Float16* __restrict__ t16,
                        const _Float16* __restrict__ Wes, const _Float16* __restrict__ Wet,
                        const float* __restrict__ bes, const float* __restrict__ bet,
                        float* __restrict__ se32, float* __restrict__ te32,
                        _Float16* __restrict__ se16, _Float16* __restrict__ te16) {
  int wid = blockIdx.x;            // 0..15
  int mat = wid >> 3, rt = wid & 7;
  const _Float16* A = (mat ? t16 : s16) + rt * 16 * DSIN;
  const _Float16* W = mat ? Wet : Wes;
  const float* bias = mat ? bet : bes;
  float*     o32 = (mat ? te32 : se32) + rt * 16 * DIM;
  _Float16*  o16 = (mat ? te16 : se16) + rt * 16 * DIM;

  v8f acc[8] = {};
  gemm16<32>(A, DSIN, W, DSIN, acc);

  int lane = threadIdx.x & 31, nloc = lane & 15, hi = lane >> 4;
#pragma unroll
  for (int tn = 0; tn < 8; ++tn) {
    int n = tn * 16 + nloc;
    float bv = bias[n];
#pragma unroll
    for (int r = 0; r < 8; ++r) {
      int m = r + hi * 8;
      float v = acc[tn][r] + bv;
      o32[m * DIM + n] = v;
      o16[m * DIM + n] = (_Float16)v;
    }
  }
}

// ---------------- 4 small projections: [128x128] x [128x128] ----------------
__global__ void k_small(const _Float16* __restrict__ se16, const _Float16* __restrict__ te16,
                        const _Float16* __restrict__ Wtv, const _Float16* __restrict__ Wtq,
                        const _Float16* __restrict__ Wtsv, const _Float16* __restrict__ Wtsq,
                        const float* __restrict__ btv, const float* __restrict__ btq,
                        const float* __restrict__ btsv, const float* __restrict__ btsq,
                        float* __restrict__ mtv, float* __restrict__ mtq,
                        float* __restrict__ mtsv, float* __restrict__ mtsqp) {
  int wid = blockIdx.x;            // 0..31
  int g = wid >> 3, rt = wid & 7;
  const _Float16* A = ((g == 3) ? se16 : te16) + rt * 16 * DIM;
  const _Float16* W = (g == 0) ? Wtv : (g == 1) ? Wtq : (g == 2) ? Wtsv : Wtsq;
  const float* bias = (g == 0) ? btv : (g == 1) ? btq : (g == 2) ? btsv : btsq;
  float* out = ((g == 0) ? mtv : (g == 1) ? mtq : (g == 2) ? mtsv : mtsqp) + rt * 16 * DIM;

  v8f acc[8] = {};
  gemm16<4>(A, DIM, W, DIM, acc);

  int lane = threadIdx.x & 31, nloc = lane & 15, hi = lane >> 4;
#pragma unroll
  for (int tn = 0; tn < 8; ++tn) {
    int n = tn * 16 + nloc;
    float bv = bias[n];
#pragma unroll
    for (int r = 0; r < 8; ++r) out[(r + hi * 8) * DIM + n] = acc[tn][r] + bv;
  }
}

// ---------------- pair chain: h = l2norm(W2 @ (W1 @ relu(V[j]-Q[i]) + b1) + b2) ----------------
// 16 pairs per wave, 8 waves per block; W1/W2 staged in LDS (async DMA).
__global__ __launch_bounds__(256) void k_pair(const float* __restrict__ V, const float* __restrict__ Q,
                                              const _Float16* __restrict__ W1g, const float* __restrict__ b1,
                                              const _Float16* __restrict__ W2g, const float* __restrict__ b2,
                                              _Float16* __restrict__ hout) {
  __shared__ _Float16 sm[2 * WMAT_H + 8 * TILE_H];
  _Float16* w1 = sm;
  _Float16* w2 = sm + WMAT_H;
  lds_copy_weight(w1, W1g);           // async, overlap with tile build below
  lds_copy_weight(w2, W2g);

  int w = threadIdx.x >> 5, lane = threadIdx.x & 31;
  int nloc = lane & 15, hi = lane >> 4;
  _Float16* tile = sm + 2 * WMAT_H + w * TILE_H;
  int p0 = (blockIdx.x * 8 + w) * 16;

  // build A tile: x[r][n] = relu(V[j][n] - Q[i][n]) for pair p0+r
  int n0 = lane * 4;
  for (int r = 0; r < 16; ++r) {
    int p = p0 + r, i = p >> 7, j = p & 127;
    v4f vv = *(const v4f*)(V + j * DIM + n0);
    v4f qq = *(const v4f*)(Q + i * DIM + n0);
#pragma unroll
    for (int c = 0; c < 4; ++c) {
      float x = vv[c] - qq[c];
      tile[r * WST + n0 + c] = (_Float16)(x > 0.f ? x : 0.f);
    }
  }
  weights_ready();

  v8f acc[8] = {};
  gemm16<4>(tile, WST, w1, WST, acc);
  __syncthreads();
#pragma unroll
  for (int tn = 0; tn < 8; ++tn) {
    float bv = b1[tn * 16 + nloc];
#pragma unroll
    for (int r = 0; r < 8; ++r)
      tile[(r + hi * 8) * WST + tn * 16 + nloc] = (_Float16)(acc[tn][r] + bv);
  }
  __syncthreads();

  v8f acc2[8] = {};
  gemm16<4>(tile, WST, w2, WST, acc2);

  float b2r[8];
#pragma unroll
  for (int tn = 0; tn < 8; ++tn) b2r[tn] = b2[tn * 16 + nloc];
#pragma unroll
  for (int tn = 0; tn < 8; ++tn)
#pragma unroll
    for (int r = 0; r < 8; ++r) acc2[tn][r] += b2r[tn];

#pragma unroll
  for (int r = 0; r < 8; ++r) {
    float ss = 0.f;
#pragma unroll
    for (int tn = 0; tn < 8; ++tn) ss += acc2[tn][r] * acc2[tn][r];
#pragma unroll
    for (int d = 1; d < 16; d <<= 1) ss += __shfl_xor(ss, d, 32);
    float rs = rsqrtf(ss);
    int p = p0 + r + hi * 8;
#pragma unroll
    for (int tn = 0; tn < 8; ++tn)
      hout[p * DIM + tn * 16 + nloc] = (_Float16)(acc2[tn][r] * rs);
  }
}

// ---------------- negatives chain + scores: one wave per (i,j) pair ----------------
__global__ __launch_bounds__(256) void k_neg(const float* __restrict__ memory_s, const int* __restrict__ idx,
                                             const float* __restrict__ mtsv,
                                             const _Float16* __restrict__ ht16, const _Float16* __restrict__ hp16,
                                             const _Float16* __restrict__ Wtsq, const float* __restrict__ btsq,
                                             const _Float16* __restrict__ Wmts, const float* __restrict__ bmts,
                                             const _Float16* __restrict__ Whts, const float* __restrict__ bhts,
                                             float* __restrict__ out) {
  __shared__ _Float16 sm[3 * WMAT_H + 8 * TILE_H];
  _Float16* w1 = sm;
  _Float16* w2 = sm + WMAT_H;
  _Float16* w3 = sm + 2 * WMAT_H;
  lds_copy_weight(w1, Wtsq);          // async DMA, overlapped with gather below
  lds_copy_weight(w2, Wmts);
  lds_copy_weight(w3, Whts);

  int w = threadIdx.x >> 5, lane = threadIdx.x & 31;
  int nloc = lane & 15, hi = lane >> 4;
  _Float16* tile = sm + 3 * WMAT_H + w * TILE_H;
  int p = blockIdx.x * 8 + w;      // pair 0..16383
  int j = p & 127;

  float vj[8], htr[8], b1r[8], b2r[8], b3r[8];
#pragma unroll
  for (int tn = 0; tn < 8; ++tn) {
    int n = tn * 16 + nloc;
    vj[tn]  = mtsv[j * DIM + n];
    htr[tn] = (float)ht16[p * DIM + n];
    b1r[tn] = btsq[n];
    b2r[tn] = bmts[n];
    b3r[tn] = bhts[n];
  }

  // gather 16 negative rows (row r = negative k) into A tile as f16
  int n0 = lane * 4;
  const int* ip = idx + p * (NNEG + 1);
  for (int r = 0; r < NNEG; ++r) {
    int id = ip[r];
    v4f mv = *(const v4f*)(memory_s + (size_t)id * DIM + n0);
#pragma unroll
    for (int c = 0; c < 4; ++c) tile[r * WST + n0 + c] = (_Float16)mv[c];
  }
  weights_ready();

  // GEMM1: q_neg = neg @ Wtsq^T + btsq ; then x = relu(vj - q_neg)
  v8f acc[8] = {};
  gemm16<4>(tile, WST, w1, WST, acc);
  __syncthreads();
#pragma unroll
  for (int tn = 0; tn < 8; ++tn)
#pragma unroll
    for (int r = 0; r < 8; ++r) {
      float x = vj[tn] - (acc[tn][r] + b1r[tn]);
      tile[(r + hi * 8) * WST + tn * 16 + nloc] = (_Float16)(x > 0.f ? x : 0.f);
    }
  __syncthreads();

  // GEMM2: r = x @ Wmts^T + bmts
  v8f acc2[8] = {};
  gemm16<4>(tile, WST, w2, WST, acc2);
  __syncthreads();
#pragma unroll
  for (int tn = 0; tn < 8; ++tn)
#pragma unroll
    for (int r = 0; r < 8; ++r)
      tile[(r + hi * 8) * WST + tn * 16 + nloc] = (_Float16)(acc2[tn][r] + b2r[tn]);
  __syncthreads();

  // GEMM3: pre_h = r @ Whts^T + bhts ; l2norm rows ; dot with h_t
  v8f acc3[8] = {};
  gemm16<4>(tile, WST, w3, WST, acc3);
#pragma unroll
  for (int tn = 0; tn < 8; ++tn)
#pragma unroll
    for (int r = 0; r < 8; ++r) acc3[tn][r] += b3r[tn];

#pragma unroll
  for (int r = 0; r < 8; ++r) {
    float ss = 0.f, sc = 0.f;
#pragma unroll
    for (int tn = 0; tn < 8; ++tn) {
      float v = acc3[tn][r];
      ss += v * v;
      sc += v * htr[tn];
    }
#pragma unroll
    for (int d = 1; d < 16; d <<= 1) {
      ss += __shfl_xor(ss, d, 32);
      sc += __shfl_xor(sc, d, 32);
    }
    float score = sc * rsqrtf(ss);
    if (nloc == 0)
      out[p * (NNEG + 1) + 1 + (r + hi * 8)] = __expf((score - 1.0f) * INV_T);
  }

  // positive score: dot(h_t, h_pos)
  float scp = 0.f;
#pragma unroll
  for (int tn = 0; tn < 8; ++tn) scp += htr[tn] * (float)hp16[p * DIM + tn * 16 + nloc];
#pragma unroll
  for (int d = 1; d < 16; d <<= 1) scp += __shfl_xor(scp, d, 32);
  if (lane == 0) out[p * (NNEG + 1)] = __expf((scp - 1.0f) * INV_T);
}

// ---------------- memory bank copy + momentum scatter update ----------------
__global__ void k_copy4(const float* __restrict__ src, float* __restrict__ dst, int n4) {
  int i = blockIdx.x * blockDim.x + threadIdx.x;
  int stride = gridDim.x * blockDim.x;
  const v4f* s4 = (const v4f*)src;
  v4f* d4 = (v4f*)dst;
  for (; i < n4; i += stride) d4[i] = s4[i];
}

__global__ void k_memupd(const float* __restrict__ memory_s, const float* __restrict__ se32,
                         const int* __restrict__ y, float* __restrict__ outmem) {
  int b = blockIdx.x, t = threadIdx.x;
  int row = y[b];
  float ab = memory_s[(size_t)row * DIM + t] * 0.5f + se32[b * DIM + t] * 0.5f;
  float ss = ab * ab;
#pragma unroll
  for (int d = 1; d < 32; d <<= 1) ss += __shfl_xor(ss, d, 32);
  __shared__ float red[4];
  if ((t & 31) == 0) red[t >> 5] = ss;
  __syncthreads();
  float tot = red[0] + red[1] + red[2] + red[3];
  outmem[(size_t)row * DIM + t] = ab * rsqrtf(tot);
}

// ---------------- host launcher ----------------
extern "C" void kernel_launch(void* const* d_in, const int* in_sizes, int n_in,
                              void* d_out, int out_size, void* d_ws, size_t ws_size,
                              hipStream_t stream) {
  const float* s        = (const float*)d_in[0];
  const float* t        = (const float*)d_in[1];
  const int*   y        = (const int*)d_in[2];
  const int*   idx      = (const int*)d_in[3];
  const float* memory_s = (const float*)d_in[4];
  const float* W_es  = (const float*)d_in[5];  const float* b_es  = (const float*)d_in[6];
  const float* W_et  = (const float*)d_in[7];  const float* b_et  = (const float*)d_in[8];
  const float* W_tv  = (const float*)d_in[9];  const float* b_tv  = (const float*)d_in[10];
  const float* W_tq  = (const float*)d_in[11]; const float* b_tq  = (const float*)d_in[12];
  const float* W_tsv = (const float*)d_in[13]; const float* b_tsv = (const float*)d_in[14];
  const float* W_tsq = (const float*)d_in[15]; const float* b_tsq = (const float*)d_in[16];
  const float* W_mt  = (const float*)d_in[17]; const float* b_mt  = (const float*)d_in[18];
  const float* W_mts = (const float*)d_in[19]; const float* b_mts = (const float*)d_in[20];
  const float* W_ht  = (const float*)d_in[21]; const float* b_ht  = (const float*)d_in[22];
  const float* W_hts = (const float*)d_in[23]; const float* b_hts = (const float*)d_in[24];

  char* ws = (char*)d_ws;
  size_t off = 0;
  auto alloc = [&](size_t bytes) -> void* {
    void* p = ws + off;
    off += (bytes + 255) & ~(size_t)255;
    return p;
  };
  _Float16* s16    = (_Float16*)alloc((size_t)BSZ * DSIN * 2);
  _Float16* t16    = (_Float16*)alloc((size_t)BSZ * DSIN * 2);
  _Float16* Wes16  = (_Float16*)alloc((size_t)DIM * DSIN * 2);
  _Float16* Wet16  = (_Float16*)alloc((size_t)DIM * DSIN * 2);
  _Float16* Wtv16  = (_Float16*)alloc((size_t)DIM * DIM * 2);
  _Float16* Wtq16  = (_Float16*)alloc((size_t)DIM * DIM * 2);
  _Float16* Wtsv16 = (_Float16*)alloc((size_t)DIM * DIM * 2);
  _Float16* Wtsq16 = (_Float16*)alloc((size_t)DIM * DIM * 2);
  _Float16* Wmt16  = (_Float16*)alloc((size_t)DIM * DIM * 2);
  _Float16* Wmts16 = (_Float16*)alloc((size_t)DIM * DIM * 2);
  _Float16* Wht16  = (_Float16*)alloc((size_t)DIM * DIM * 2);
  _Float16* Whts16 = (_Float16*)alloc((size_t)DIM * DIM * 2);
  float*    se32   = (float*)alloc((size_t)BSZ * DIM * 4);
  float*    te32   = (float*)alloc((size_t)BSZ * DIM * 4);
  _Float16* se16   = (_Float16*)alloc((size_t)BSZ * DIM * 2);
  _Float16* te16   = (_Float16*)alloc((size_t)BSZ * DIM * 2);
  float*    mtv    = (float*)alloc((size_t)BSZ * DIM * 4);
  float*    mtq    = (float*)alloc((size_t)BSZ * DIM * 4);
  float*    mtsv   = (float*)alloc((size_t)BSZ * DIM * 4);
  float*    mtsqp  = (float*)alloc((size_t)BSZ * DIM * 4);
  _Float16* ht16   = (_Float16*)alloc((size_t)NPAIR * DIM * 2);
  _Float16* hp16   = (_Float16*)alloc((size_t)NPAIR * DIM * 2);

  float* outp   = (float*)d_out;
  float* outmem = outp + (size_t)NPAIR * (NNEG + 1);

  auto cvt = [&](const float* src, _Float16* dst, int n) {
    k_cvt<<<(n + 255) / 256, 256, 0, stream>>>(src, dst, n);
  };
  cvt(s, s16, BSZ * DSIN);
  cvt(t, t16, BSZ * DSIN);
  cvt(W_es, Wes16, DIM * DSIN);
  cvt(W_et, Wet16, DIM * DSIN);
  cvt(W_tv, Wtv16, DIM * DIM);
  cvt(W_tq, Wtq16, DIM * DIM);
  cvt(W_tsv, Wtsv16, DIM * DIM);
  cvt(W_tsq, Wtsq16, DIM * DIM);
  cvt(W_mt, Wmt16, DIM * DIM);
  cvt(W_mts, Wmts16, DIM * DIM);
  cvt(W_ht, Wht16, DIM * DIM);
  cvt(W_hts, Whts16, DIM * DIM);

  k_embed<<<16, 32, 0, stream>>>(s16, t16, Wes16, Wet16, b_es, b_et, se32, te32, se16, te16);
  k_small<<<32, 32, 0, stream>>>(se16, te16, Wtv16, Wtq16, Wtsv16, Wtsq16,
                                 b_tv, b_tq, b_tsv, b_tsq, mtv, mtq, mtsv, mtsqp);
  // h_t: relu(mtv[j]-mtq[i]) -> W_mt -> W_ht -> l2norm
  k_pair<<<NPAIR / 128, 256, 0, stream>>>(mtv, mtq, Wmt16, b_mt, Wht16, b_ht, ht16);
  // h_pos: relu(mtsv[j]-mtsqp[i]) -> W_mts -> W_hts -> l2norm
  k_pair<<<NPAIR / 128, 256, 0, stream>>>(mtsv, mtsqp, Wmts16, b_mts, Whts16, b_hts, hp16);
  // negatives + all 17 scores per pair
  k_neg<<<NPAIR / 8, 256, 0, stream>>>(memory_s, idx, mtsv, ht16, hp16,
                                       Wtsq16, b_tsq, Wmts16, b_mts, Whts16, b_hts, outp);
  // memory bank: copy then momentum-normalized scatter
  k_copy4<<<2048, 256, 0, stream>>>(memory_s, outmem, NMEM * DIM / 4);
  k_memupd<<<BSZ, DIM, 0, stream>>>(memory_s, se32, y, outmem);
}
